// GRU_Decoder_2_63754494542448
// MI455X (gfx1250) — matrix operand order repaired and checked by reading, not compiled, not run on hardware
//
#include <hip/hip_runtime.h>
#include <hip/hip_bf16.h>

#define B_     32
#define H_     1024
#define L_     3
#define V_     32000
#define T_     128
#define K_     1024
#define G3_    (3 * H_)

typedef __attribute__((ext_vector_type(16))) __bf16 v16bf;
typedef __attribute__((ext_vector_type(8)))  __bf16 v8bf;
typedef __attribute__((ext_vector_type(8)))  float  v8f;

__device__ __forceinline__ __bf16 f2bf(float f) {
  unsigned u = __float_as_uint(f);
  unsigned r = (u >> 16) & 1u;          // round-to-nearest-even
  u += 0x7FFFu + r;
  unsigned short h = (unsigned short)(u >> 16);
  return __builtin_bit_cast(__bf16, h);
}

__device__ __forceinline__ float sigm(float x) {
  return 1.0f / (1.0f + __expf(-x));
}

__device__ __forceinline__ unsigned long long amax_key(float v, int col) {
  unsigned ub  = __float_as_uint(v);
  unsigned ord = (ub & 0x80000000u) ? ~ub : (ub | 0x80000000u); // order-preserving
  return ((unsigned long long)ord << 32) | (unsigned)(~col);    // ties -> lowest col
}

// ---------------- f32 -> bf16 convert (grid-stride) ----------------
__global__ void cvt_bf16_kernel(const float* __restrict__ src,
                                __bf16* __restrict__ dst, int n) {
  int i = blockIdx.x * blockDim.x + threadIdx.x;
  int stride = gridDim.x * blockDim.x;
  for (; i < n; i += stride) dst[i] = f2bf(src[i]);
}

// ---------------- dual-job WMMA GEMM: C[32,N] = A[32,1024] * W[N,1024]^T + bias
// blockDim.x = 128 (4 waves); each wave: one 32x16 tile (two 16x16 f32 accums)
__global__ void __launch_bounds__(128)
wmma_mm_kernel(const __bf16* __restrict__ A0, const __bf16* __restrict__ W0,
               const float* __restrict__ b0, float* __restrict__ C0,
               const __bf16* __restrict__ A1, const __bf16* __restrict__ W1,
               const float* __restrict__ b1, float* __restrict__ C1, int N) {
  const __bf16* A    = blockIdx.y ? A1 : A0;
  const __bf16* W    = blockIdx.y ? W1 : W0;
  const float*  bias = blockIdx.y ? b1 : b0;
  float*        C    = blockIdx.y ? C1 : C0;

  const int wave  = threadIdx.x >> 5;
  const int lane  = threadIdx.x & 31;
  const int lhalf = lane >> 4;
  const int lrow  = lane & 15;
  const int ncol  = (blockIdx.x * 4 + wave) * 16;
  if (ncol >= N) return;

  union Frag { v16bf v; v8bf h[2]; };
  v8f c0 = {}, c1 = {};

  // A fragment (16-bit A 16x32 layout): lane<16 -> K {0..7,16..23}, lane>=16 -> K {8..15,24..31}
  const __bf16* arow0 = A + (size_t)lrow * K_ + lhalf * 8;
  const __bf16* arow1 = arow0 + (size_t)16 * K_;
  // B fragment (32x16): lane<16 -> K {0..7,16..23} of column lrow, lane>=16 -> K {8..15,24..31}
  const __bf16* wrow  = W + (size_t)(ncol + lrow) * K_ + lhalf * 8;

  for (int kb = 0; kb < K_; kb += 32) {
    Frag a0, a1, b;
    a0.h[0] = *(const v8bf*)(arow0 + kb);
    a0.h[1] = *(const v8bf*)(arow0 + kb + 16);
    a1.h[0] = *(const v8bf*)(arow1 + kb);
    a1.h[1] = *(const v8bf*)(arow1 + kb + 16);
    b.h[0]  = *(const v8bf*)(wrow + kb);
    b.h[1]  = *(const v8bf*)(wrow + kb + 16);
    __builtin_prefetch(wrow + kb + 256, 0, 3);   // WGP-scope: pull into near caches
    c0 = __builtin_amdgcn_wmma_f32_16x16x32_bf16(false, a0.v, false, b.v,
                                                 (short)0, c0, false, false);
    c1 = __builtin_amdgcn_wmma_f32_16x16x32_bf16(false, a1.v, false, b.v,
                                                 (short)0, c1, false, false);
  }

  const int col = ncol + lrow;
  const float bv = bias ? bias[col] : 0.0f;
#pragma unroll
  for (int r = 0; r < 8; ++r) {
    int row = r + lhalf * 8;                  // C layout: vgpr r -> M=r / M=r+8
    C[(size_t)row * N + col]        = c0[r] + bv;
    C[(size_t)(row + 16) * N + col] = c1[r] + bv;
  }
}

// ---------------- FC + argmax: logits -> d_out[b,t,:], packed-key max -> amax[32]
// Each wave: 32x32 tile (4 accums) -> A fragments amortized over two N-tiles.
__global__ void __launch_bounds__(128)
wmma_fc_argmax_kernel(const __bf16* __restrict__ A, const __bf16* __restrict__ W,
                      const float* __restrict__ bias, float* __restrict__ out,
                      unsigned long long* __restrict__ amax, int t) {
  __shared__ unsigned long long smax[B_];
  if (threadIdx.x < B_) smax[threadIdx.x] = 0ull;
  __syncthreads();

  const int wave  = threadIdx.x >> 5;
  const int lane  = threadIdx.x & 31;
  const int lhalf = lane >> 4;
  const int lrow  = lane & 15;
  const int ncol  = (blockIdx.x * 4 + wave) * 32;   // 32 cols per wave

  union Frag { v16bf v; v8bf h[2]; };
  v8f c00 = {}, c01 = {}, c10 = {}, c11 = {};
  const __bf16* arow0 = A + (size_t)lrow * K_ + lhalf * 8;
  const __bf16* arow1 = arow0 + (size_t)16 * K_;
  const __bf16* wrow0 = W + (size_t)(ncol + lrow) * K_ + lhalf * 8;
  const __bf16* wrow1 = wrow0 + (size_t)16 * K_;

  for (int kb = 0; kb < K_; kb += 32) {
    Frag a0, a1, b0, b1;
    a0.h[0] = *(const v8bf*)(arow0 + kb);
    a0.h[1] = *(const v8bf*)(arow0 + kb + 16);
    a1.h[0] = *(const v8bf*)(arow1 + kb);
    a1.h[1] = *(const v8bf*)(arow1 + kb + 16);
    b0.h[0] = *(const v8bf*)(wrow0 + kb);
    b0.h[1] = *(const v8bf*)(wrow0 + kb + 16);
    b1.h[0] = *(const v8bf*)(wrow1 + kb);
    b1.h[1] = *(const v8bf*)(wrow1 + kb + 16);
    __builtin_prefetch(wrow0 + kb + 256, 0, 3);
    __builtin_prefetch(wrow1 + kb + 256, 0, 3);
    c00 = __builtin_amdgcn_wmma_f32_16x16x32_bf16(false, a0.v, false, b0.v,
                                                  (short)0, c00, false, false);
    c10 = __builtin_amdgcn_wmma_f32_16x16x32_bf16(false, a1.v, false, b0.v,
                                                  (short)0, c10, false, false);
    c01 = __builtin_amdgcn_wmma_f32_16x16x32_bf16(false, a0.v, false, b1.v,
                                                  (short)0, c01, false, false);
    c11 = __builtin_amdgcn_wmma_f32_16x16x32_bf16(false, a1.v, false, b1.v,
                                                  (short)0, c11, false, false);
  }

  const float bv0 = bias[ncol + lrow];
  const float bv1 = bias[ncol + 16 + lrow];
  float* op0 = out + (size_t)t * V_ + ncol + lrow;
  float* op1 = op0 + 16;
#pragma unroll
  for (int r = 0; r < 8; ++r) {
    int row = r + lhalf * 8;
    float v00 = c00[r] + bv0, v10 = c10[r] + bv0;
    float v01 = c01[r] + bv1, v11 = c11[r] + bv1;
    op0[(size_t)row * ((size_t)T_ * V_)]        = v00;
    op0[(size_t)(row + 16) * ((size_t)T_ * V_)] = v10;
    op1[(size_t)row * ((size_t)T_ * V_)]        = v01;
    op1[(size_t)(row + 16) * ((size_t)T_ * V_)] = v11;
    atomicMax(&smax[row],      amax_key(v00, ncol + lrow));       // ds_max_u64
    atomicMax(&smax[row + 16], amax_key(v10, ncol + lrow));
    atomicMax(&smax[row],      amax_key(v01, ncol + 16 + lrow));
    atomicMax(&smax[row + 16], amax_key(v11, ncol + 16 + lrow));
  }
  __syncthreads();
  if (threadIdx.x < B_)
    atomicMax(&amax[threadIdx.x], smax[threadIdx.x]); // global_atomic_max_u64
}

// ---------------- embedding gather (tokens from prev-step argmax keys) --------
__global__ void embed_kernel(const unsigned long long* __restrict__ amax,
                             const __bf16* __restrict__ embb,
                             __bf16* __restrict__ xb, int t) {
  int b = blockIdx.x;
  int tok = 0;
  if (t != 0) tok = (int)(~(unsigned)(amax[b] & 0xffffffffull));
  const unsigned long long* src =
      (const unsigned long long*)(embb + (size_t)tok * H_);
  unsigned long long* dst = (unsigned long long*)(xb + (size_t)b * H_);
  dst[threadIdx.x] = src[threadIdx.x];   // 256 thr * 8B = 1024 bf16 row
}

// ---------------- GRU gate fuse: h = (1-z)*n + z*h ; refresh bf16 copies ------
__global__ void gru_gate_kernel(const float* __restrict__ gx,
                                const float* __restrict__ gh,
                                float* __restrict__ hf,       // h[l], in/out
                                __bf16* __restrict__ hb,      // h[l] bf16
                                __bf16* __restrict__ xb,      // next layer input
                                unsigned long long* __restrict__ amax,
                                int doreset) {
  int idx = blockIdx.x * blockDim.x + threadIdx.x;  // 0 .. B*H-1
  int b = idx >> 10, j = idx & (H_ - 1);
  const float* gxr = gx + (size_t)b * G3_;
  const float* ghr = gh + (size_t)b * G3_;
  float r  = sigm(gxr[j] + ghr[j]);
  float zg = sigm(gxr[H_ + j] + ghr[H_ + j]);
  float ni = gxr[2 * H_ + j] + r * ghr[2 * H_ + j];
  float e2 = __expf(-2.0f * ni);
  float n  = (1.0f - e2) / (1.0f + e2);             // tanh
  float h  = (1.0f - zg) * n + zg * hf[idx];
  hf[idx] = h;
  __bf16 hv = f2bf(h);
  hb[idx] = hv;
  xb[idx] = hv;
  if (doreset && idx < B_) amax[idx] = 0ull;        // before this step's FC
}

extern "C" void kernel_launch(void* const* d_in, const int* in_sizes, int n_in,
                              void* d_out, int out_size, void* d_ws, size_t ws_size,
                              hipStream_t stream) {
  (void)in_sizes; (void)n_in; (void)out_size; (void)ws_size;
  const float* z    = (const float*)d_in[0];
  const float* emb  = (const float*)d_in[1];
  const float* wlat = (const float*)d_in[2];
  const float* blat = (const float*)d_in[3];
  const float* wih  = (const float*)d_in[4];
  const float* whh  = (const float*)d_in[5];
  const float* bih  = (const float*)d_in[6];
  const float* bhh  = (const float*)d_in[7];
  const float* fcw  = (const float*)d_in[8];
  const float* fcb  = (const float*)d_in[9];
  float* out = (float*)d_out;

  char* wp = (char*)d_ws;
  auto alloc = [&](size_t bytes) {
    char* p = wp; wp += (bytes + 255) & ~(size_t)255; return p;
  };
  __bf16* emb_b  = (__bf16*)alloc((size_t)V_ * K_ * 2);
  __bf16* fcw_b  = (__bf16*)alloc((size_t)V_ * K_ * 2);
  __bf16* wih_b  = (__bf16*)alloc((size_t)L_ * G3_ * K_ * 2);
  __bf16* whh_b  = (__bf16*)alloc((size_t)L_ * G3_ * K_ * 2);
  __bf16* wlat_b = (__bf16*)alloc((size_t)G3_ * K_ * 2);
  __bf16* z_b    = (__bf16*)alloc((size_t)B_ * K_ * 2);
  float*  gx     = (float*)alloc((size_t)B_ * G3_ * 4);
  float*  gh     = (float*)alloc((size_t)B_ * G3_ * 4);
  float*  hf     = (float*)alloc((size_t)L_ * B_ * H_ * 4);
  __bf16* hb     = (__bf16*)alloc((size_t)L_ * B_ * H_ * 2);
  __bf16* xb     = (__bf16*)alloc((size_t)B_ * H_ * 2);
  unsigned long long* amax = (unsigned long long*)alloc(B_ * 8);

  auto cvt = [&](const float* s, __bf16* d, size_t n) {
    int blocks = (int)((n + 255) / 256);
    if (blocks > 8192) blocks = 8192;
    cvt_bf16_kernel<<<blocks, 256, 0, stream>>>(s, d, (int)n);
  };
  cvt(emb,  emb_b,  (size_t)V_ * K_);
  cvt(fcw,  fcw_b,  (size_t)V_ * K_);
  cvt(wih,  wih_b,  (size_t)L_ * G3_ * K_);
  cvt(whh,  whh_b,  (size_t)L_ * G3_ * K_);
  cvt(wlat, wlat_b, (size_t)G3_ * K_);
  cvt(z,    z_b,    (size_t)B_ * K_);

  // Initial hidden: [B, L*H] row-major == flat(L,B,H) reshape => write into hf.
  wmma_mm_kernel<<<dim3(G3_ / 64, 1), 128, 0, stream>>>(
      z_b, wlat_b, blat, hf, z_b, wlat_b, blat, hf, G3_);
  cvt_bf16_kernel<<<(L_ * B_ * H_) / 256, 256, 0, stream>>>(hf, hb, L_ * B_ * H_);

  for (int t = 0; t < T_; ++t) {
    embed_kernel<<<B_, 256, 0, stream>>>(amax, emb_b, xb, t);
    for (int l = 0; l < L_; ++l) {
      wmma_mm_kernel<<<dim3(G3_ / 64, 2), 128, 0, stream>>>(
          xb,                       wih_b + (size_t)l * G3_ * K_,
          bih + (size_t)l * G3_,    gx,
          hb + (size_t)l * B_ * H_, whh_b + (size_t)l * G3_ * K_,
          bhh + (size_t)l * G3_,    gh, G3_);
      gru_gate_kernel<<<(B_ * H_) / 256, 256, 0, stream>>>(
          gx, gh, hf + (size_t)l * B_ * H_, hb + (size_t)l * B_ * H_,
          xb, amax, (l == 0) ? 1 : 0);
    }
    wmma_fc_argmax_kernel<<<V_ / 128, 128, 0, stream>>>(
        xb, fcw_b, fcb, out, amax, t);
  }
}
